// AttLayer_704374636900
// MI455X (gfx1250) — matrix operand order repaired and checked by reading, not compiled, not run on hardware
//
#include <hip/hip_runtime.h>
#include <hip/hip_bf16.h>

// Sliding-window attention for MI455X (gfx1250, wave32, WMMA + async-to-LDS).
// All GEMMs run on v_wmma_f32_16x16x32_f16 (f16 inputs, fp32 accum); all f16
// tile staging uses GLOBAL_LOAD_ASYNC_TO_LDS_B128 (ASYNCcnt) when available.
//
// Workspace layout (requires 256 MiB of d_ws):
//   [0]        qkT  f16 [32768][512]   (cols 0-255 = q, 256-511 = k, l-major)
//   [+32 MiB]  vh   f16 [256][32768]   (v, c-major)
//   [+48 MiB]  ener f32 [64*512][1024] (scaled energy)
//   [+176MiB]  attn f16 [64*512][1024]
//   [+240MiB]  uT   f16 [32768][256]   (post-ReLU attention output, l-major)

#define L_LEN   32768
#define CDIM    256
#define QDIM    512
#define NB      64
#define BLQ     512
#define HALFW   256
#define WIN     1024

typedef __attribute__((ext_vector_type(16))) _Float16 v16h;
typedef __attribute__((ext_vector_type(8)))  float    v8f;
typedef _Float16 h16;

struct __align__(16) u4x { unsigned int x, y, z, w; };
struct __align__(16) f4x { float x, y, z, w; };

union H2u { unsigned int u; h16 h[2]; };
union H8u { u4x q; h16 h[8]; };
union FrU { v16h v; unsigned int u[8]; };

// ---------------- async global -> LDS copy (CDNA5 path) ---------------------
#if __has_builtin(__builtin_amdgcn_global_load_async_to_lds_b128)
#define HAVE_ASYNC_LDS 1
typedef int v4i_t __attribute__((vector_size(16)));
typedef __attribute__((address_space(1))) v4i_t g_v4i;
typedef __attribute__((address_space(3))) v4i_t l_v4i;
// Copy 16 bytes global->LDS via the async DMA path (no VGPR roundtrip).
__device__ __forceinline__ void cp16(h16* dst, const h16* src) {
  __builtin_amdgcn_global_load_async_to_lds_b128(
      (g_v4i*)(uintptr_t)src,
      (l_v4i*)(unsigned int)(uintptr_t)dst, 0, 0);
}
__device__ __forceinline__ void async_join() {
#if __has_builtin(__builtin_amdgcn_s_wait_asynccnt)
  __builtin_amdgcn_s_wait_asynccnt(0);
#else
  asm volatile("s_wait_asynccnt 0x0" ::: "memory");
#endif
}
#else
__device__ __forceinline__ void cp16(h16* dst, const h16* src) {
  *(u4x*)dst = *(const u4x*)src;
}
__device__ __forceinline__ void async_join() {}
#endif

// ---------------- WMMA helpers ---------------------------------------------
__device__ __forceinline__ v8f wmma16(v16h a, v16h b, v8f c) {
  return __builtin_amdgcn_wmma_f32_16x16x32_f16(false, a, false, b, (short)0, c,
                                                false, false);
}

// A fragment (16x32 f16) from LDS laid out [m][k], row stride sh halves.
// lanes 0-15/16-31 hold M=lane&15; elems 0-7 = k=hi*8+e, 8-15 = k=16+hi*8+e-8.
__device__ __forceinline__ v16h afrag(const h16* base, int sh) {
  int lane = threadIdx.x & 31;
  int m = lane & 15, hi = lane >> 4;
  const unsigned int* p0 = (const unsigned int*)(base + m * sh + hi * 8);
  const unsigned int* p1 = (const unsigned int*)(base + m * sh + hi * 8 + 16);
  FrU f;
#pragma unroll
  for (int i = 0; i < 4; ++i) { f.u[i] = p0[i]; f.u[4 + i] = p1[i]; }
  return f.v;
}

// B fragment (32x16 f16) from LDS laid out TRANSPOSED [n][k], stride sh halves.
__device__ __forceinline__ v16h bfrag(const h16* base, int sh) {
  int lane = threadIdx.x & 31;
  int n = lane & 15, hi = lane >> 4;
  const unsigned int* p0 = (const unsigned int*)(base + n * sh + hi * 8);
  const unsigned int* p1 = (const unsigned int*)(base + n * sh + hi * 8 + 16);
  FrU f;
#pragma unroll
  for (int i = 0; i < 4; ++i) { f.u[i] = p0[i]; f.u[4 + i] = p1[i]; }
  return f.v;
}

// Convert 16 contiguous fp32 to f16 and store 32B into LDS (16B aligned).
__device__ __forceinline__ void stage_a_f32(const float* src, h16* dst) {
  f4x f0 = *(const f4x*)(src + 0);
  f4x f1 = *(const f4x*)(src + 4);
  f4x f2 = *(const f4x*)(src + 8);
  f4x f3 = *(const f4x*)(src + 12);
  H8u o0, o1;
  o0.h[0] = (h16)f0.x; o0.h[1] = (h16)f0.y; o0.h[2] = (h16)f0.z; o0.h[3] = (h16)f0.w;
  o0.h[4] = (h16)f1.x; o0.h[5] = (h16)f1.y; o0.h[6] = (h16)f1.z; o0.h[7] = (h16)f1.w;
  o1.h[0] = (h16)f2.x; o1.h[1] = (h16)f2.y; o1.h[2] = (h16)f2.z; o1.h[3] = (h16)f2.w;
  o1.h[4] = (h16)f3.x; o1.h[5] = (h16)f3.y; o1.h[6] = (h16)f3.z; o1.h[7] = (h16)f3.w;
  *(u4x*)(dst) = o0.q;
  *(u4x*)(dst + 8) = o1.q;
}

// ---------------------------------------------------------------------------
// K1: fused QKV projection, M=768, N=32768, K=512 (32-steps), tile 128x128.
// q,k written transposed into qkT[l][512]; v written naturally vh[c][l].
// grid = (L/128, 6), 256 threads.
// ---------------------------------------------------------------------------
__global__ __launch_bounds__(256) void k1_qkv(
    const float* __restrict__ x1,
    const float* __restrict__ Wq, const float* __restrict__ bq,
    const float* __restrict__ Wk, const float* __restrict__ bk,
    const float* __restrict__ Wv, const float* __restrict__ bv,
    h16* __restrict__ qkT, h16* __restrict__ vh) {
  __shared__ h16 As[128 * 40];   // [m][k] W tile (converted fp32->f16)
  __shared__ h16 Bt[128 * 40];   // [n][k] transposed x1 tile
  const int t = threadIdx.x;
  const int nb0 = blockIdx.x * 128;
  const int mb0 = blockIdx.y * 128;
  const int wid = t >> 5, lane = t & 31;

  v8f acc[8];
#pragma unroll
  for (int j = 0; j < 8; ++j)
#pragma unroll
    for (int r = 0; r < 8; ++r) acc[j][r] = 0.0f;

  const int arow = t >> 1, ah = (t & 1) * 16;
  const int gmr = mb0 + arow;
  const float* Wrow = (gmr < 256) ? (Wq + (size_t)gmr * QDIM)
                    : (gmr < 512) ? (Wk + (size_t)(gmr - 256) * QDIM)
                                  : (Wv + (size_t)(gmr - 512) * QDIM);
  const int kk  = (t >> 4) << 1;
  const int nn0 = (t & 15) * 8;

  for (int kb = 0; kb < QDIM; kb += 32) {
    stage_a_f32(Wrow + kb + ah, &As[arow * 40 + ah]);
    {
      const float* r0 = x1 + (size_t)(kb + kk) * L_LEN + nb0 + nn0;
      const float* r1 = r0 + L_LEN;
      f4x a0 = *(const f4x*)(r0), a1 = *(const f4x*)(r0 + 4);
      f4x b0 = *(const f4x*)(r1), b1 = *(const f4x*)(r1 + 4);
      float fa[8] = {a0.x, a0.y, a0.z, a0.w, a1.x, a1.y, a1.z, a1.w};
      float fb[8] = {b0.x, b0.y, b0.z, b0.w, b1.x, b1.y, b1.z, b1.w};
#pragma unroll
      for (int i = 0; i < 8; ++i) {
        H2u p; p.h[0] = (h16)fa[i]; p.h[1] = (h16)fb[i];
        *(unsigned int*)&Bt[(nn0 + i) * 40 + kk] = p.u;
      }
    }
    __syncthreads();
    v16h a = afrag(As + (wid * 16) * 40, 40);
#pragma unroll
    for (int j = 0; j < 8; ++j) {
      v16h b = bfrag(Bt + (j * 16) * 40, 40);
      acc[j] = wmma16(a, b, acc[j]);
    }
    __syncthreads();
  }

  const int col = lane & 15, hi = lane >> 4;
#pragma unroll
  for (int j = 0; j < 8; ++j)
#pragma unroll
    for (int r = 0; r < 8; ++r) {
      int gm = mb0 + wid * 16 + hi * 8 + r;
      int gl = nb0 + j * 16 + col;
      float bias = (gm < 256) ? bq[gm] : (gm < 512) ? bk[gm - 256] : bv[gm - 512];
      h16 val = (h16)(acc[j][r] + bias);
      if (gm < 512) qkT[(size_t)gl * QDIM + gm] = val;      // q|k transposed
      else          vh[(size_t)(gm - 512) * L_LEN + gl] = val;
    }
}

// ---------------------------------------------------------------------------
// K2: energy[l][m] = (1/16) * sum_c q[c][l]*k[c][gm], gm = n*512-256+m (0 OOB).
// q^T and k^T tiles are contiguous rows of qkT -> async copies to LDS.
// grid = (8, 64), 256 threads.
// ---------------------------------------------------------------------------
__global__ __launch_bounds__(256) void k2_energy(
    const h16* __restrict__ qkT, float* __restrict__ ener) {
  __shared__ h16 Aq[64 * 264];   // [l][c] q tile
  __shared__ h16 Bk[32 * 264];   // [m][c] k window tile
  const int t = threadIdx.x;
  const int lc = blockIdx.x;
  const int n  = blockIdx.y;
  const int l0 = n * BLQ + lc * 64;
  const int wid = t >> 5, lane = t & 31;

  {  // Aq: 64 rows x 256 halves, pure copy from qkT rows (cols 0..255)
    const int row = t >> 2;
    const int q4  = (t & 3) * 64;
    const h16* src = qkT + (size_t)(l0 + row) * QDIM + q4;
    h16* dst = &Aq[row * 264 + q4];
#pragma unroll
    for (int i = 0; i < 8; ++i) cp16(dst + i * 8, src + i * 8);
  }
  async_join();
  __syncthreads();

  const int tl = wid & 3, tm = wid >> 2;
  const int gm_base = n * BLQ - HALFW;
  const float scale = 0.0625f;  // 1/sqrt(256)

  for (int mc = 0; mc < WIN; mc += 32) {
    {  // Bk: 32 rows x 256 halves from qkT rows (cols 256..511); zero OOB rows
      const int m  = t >> 3;
      const int o8 = (t & 7) * 32;
      const int gm = gm_base + mc + m;
      h16* dst = &Bk[m * 264 + o8];
      if (gm >= 0 && gm < L_LEN) {
        const h16* src = qkT + (size_t)gm * QDIM + 256 + o8;
#pragma unroll
        for (int i = 0; i < 4; ++i) cp16(dst + i * 8, src + i * 8);
      } else {
        const u4x z = {0u, 0u, 0u, 0u};
#pragma unroll
        for (int i = 0; i < 4; ++i) *(u4x*)(dst + i * 8) = z;
      }
    }
    async_join();
    __syncthreads();

    v8f acc;
#pragma unroll
    for (int r = 0; r < 8; ++r) acc[r] = 0.0f;
#pragma unroll
    for (int ks = 0; ks < 8; ++ks) {
      v16h a = afrag(Aq + (tl * 16) * 264 + ks * 32, 264);
      v16h b = bfrag(Bk + (tm * 16) * 264 + ks * 32, 264);
      acc = wmma16(a, b, acc);
    }
    const int col = lane & 15, hi = lane >> 4;
    const int mout = mc + tm * 16 + col;
    float* ebase = ener + ((size_t)n * BLQ + lc * 64) * WIN;
#pragma unroll
    for (int r = 0; r < 8; ++r) {
      int lrow = tl * 16 + hi * 8 + r;
      ebase[(size_t)lrow * WIN + mout] = acc[r] * scale;
    }
    __syncthreads();
  }
}

// ---------------------------------------------------------------------------
// K3: row softmax with log-mask offsets (faithful to reference).
// One wave per 1024-wide row; grid = 64*512/8, 256 threads.
// ---------------------------------------------------------------------------
__global__ __launch_bounds__(256) void k3_softmax(
    const float* __restrict__ ener, const float* __restrict__ mask,
    h16* __restrict__ attn) {
  const int wid = threadIdx.x >> 5, lane = threadIdx.x & 31;
  const int row = blockIdx.x * 8 + wid;      // n*512 + l
  const int n = row >> 9, l = row & 511;
  const float* er = ener + (size_t)row * WIN;
  h16* ar = attn + (size_t)row * WIN;
  const int gk0 = n * BLQ - HALFW;

  float ev[32], fm[32];
  float mx = -3.0e38f;
#pragma unroll
  for (int i = 0; i < 32; ++i) {
    int m = i * 32 + lane;
    int gk = gk0 + m;
    bool inw = (m >= l) && (m < l + BLQ);
    float pm = (gk >= 0 && gk < L_LEN) ? mask[gk] : 0.0f;
    float f = inw ? pm : 0.0f;
    fm[i] = f;
    ev[i] = er[m] + __logf(f + 1e-6f);
    mx = fmaxf(mx, ev[i]);
  }
#pragma unroll
  for (int off = 16; off >= 1; off >>= 1) mx = fmaxf(mx, __shfl_xor(mx, off, 32));
  float s = 0.0f;
#pragma unroll
  for (int i = 0; i < 32; ++i) { ev[i] = __expf(ev[i] - mx); s += ev[i]; }
#pragma unroll
  for (int off = 16; off >= 1; off >>= 1) s += __shfl_xor(s, off, 32);
  const float inv = 1.0f / s;
#pragma unroll
  for (int i = 0; i < 32; ++i) {
    int m = i * 32 + lane;
    ar[m] = (h16)(ev[i] * inv * fm[i]);
  }
}

// ---------------------------------------------------------------------------
// K4: u[c][l] = relu( sum_m v[c][gm] * attn[l][m] ),  gm = n*512-256+m.
// Both tiles are pure copies -> async-to-LDS. Output stored transposed uT[l][c].
// Tile 128(c) x 128(l), K=1024 in 32-steps. grid = (4, 2, 64), 256 threads.
// ---------------------------------------------------------------------------
__global__ __launch_bounds__(256) void k4_attout(
    const h16* __restrict__ vh, const h16* __restrict__ attn,
    h16* __restrict__ uT) {
  __shared__ h16 Av[128 * 40];   // v tile [c][m]
  __shared__ h16 Bt[128 * 40];   // attn tile [l][m] (natural B-transposed)
  const int t = threadIdx.x;
  const int lb = blockIdx.x, cb = blockIdx.y, n = blockIdx.z;
  const int wid = t >> 5, lane = t & 31;
  const int gl0 = n * BLQ + lb * 128;
  const int c0  = cb * 128;
  const int gmb = n * BLQ - HALFW;

  v8f acc[8];
#pragma unroll
  for (int j = 0; j < 8; ++j)
#pragma unroll
    for (int r = 0; r < 8; ++r) acc[j][r] = 0.0f;

  const int arow = t >> 1, ah = (t & 1) * 16;
  const h16* vbase = vh + (size_t)(c0 + arow) * L_LEN;
  const h16* abase = attn + ((size_t)n * BLQ + lb * 128 + arow) * WIN;

  for (int kb = 0; kb < WIN; kb += 32) {
    {  // v tile (zero outside [0,L): matches reference zero padding)
      const int gs = gmb + kb + ah;
      h16* dst = &Av[arow * 40 + ah];
      if (gs >= 0 && gs + 16 <= L_LEN) {
        cp16(dst, vbase + gs);
        cp16(dst + 8, vbase + gs + 8);
      } else {
        H8u o0, o1;
#pragma unroll
        for (int i = 0; i < 8; ++i) {
          int g0 = gs + i, g1 = gs + 8 + i;
          o0.h[i] = (g0 >= 0 && g0 < L_LEN) ? vbase[g0] : (h16)0.0f;
          o1.h[i] = (g1 >= 0 && g1 < L_LEN) ? vbase[g1] : (h16)0.0f;
        }
        *(u4x*)dst = o0.q;
        *(u4x*)(dst + 8) = o1.q;
      }
    }
    {  // attn tile: pure async copy
      h16* dst = &Bt[arow * 40 + ah];
      cp16(dst, abase + kb + ah);
      cp16(dst + 8, abase + kb + ah + 8);
    }
    async_join();
    __syncthreads();
    v16h a = afrag(Av + (wid * 16) * 40, 40);
#pragma unroll
    for (int j = 0; j < 8; ++j) {
      v16h b = bfrag(Bt + (j * 16) * 40, 40);
      acc[j] = wmma16(a, b, acc[j]);
    }
    __syncthreads();
  }

  const int col = lane & 15, hi = lane >> 4;
#pragma unroll
  for (int j = 0; j < 8; ++j)
#pragma unroll
    for (int r = 0; r < 8; ++r) {
      int c  = c0 + wid * 16 + hi * 8 + r;
      int gl = gl0 + j * 16 + col;
      float vv = acc[j][r];
      uT[(size_t)gl * CDIM + c] = (h16)(vv > 0.0f ? vv : 0.0f);
    }
}

// ---------------------------------------------------------------------------
// K5: out[o][l] = (sum_c Wo[o][c]*u[c][l] + bo[o]) * mask[l].
// B tile = rows of uT -> async copy. Tile 128x128, K=256. grid = (L/128, 4).
// ---------------------------------------------------------------------------
__global__ __launch_bounds__(256) void k5_outproj(
    const h16* __restrict__ uT, const float* __restrict__ Wo,
    const float* __restrict__ bo, const float* __restrict__ mask,
    float* __restrict__ out) {
  __shared__ h16 As[128 * 40];
  __shared__ h16 Bt[128 * 40];
  const int t = threadIdx.x;
  const int nb0 = blockIdx.x * 128;
  const int mb0 = blockIdx.y * 128;
  const int wid = t >> 5, lane = t & 31;

  v8f acc[8];
#pragma unroll
  for (int j = 0; j < 8; ++j)
#pragma unroll
    for (int r = 0; r < 8; ++r) acc[j][r] = 0.0f;

  const int arow = t >> 1, ah = (t & 1) * 16;
  const float* Wrow = Wo + (size_t)(mb0 + arow) * CDIM;
  const h16* ubase = uT + (size_t)(nb0 + arow) * CDIM;

  for (int kb = 0; kb < CDIM; kb += 32) {
    stage_a_f32(Wrow + kb + ah, &As[arow * 40 + ah]);
    {
      h16* dst = &Bt[arow * 40 + ah];
      cp16(dst, ubase + kb + ah);
      cp16(dst + 8, ubase + kb + ah + 8);
    }
    async_join();
    __syncthreads();
    v16h a = afrag(As + (wid * 16) * 40, 40);
#pragma unroll
    for (int j = 0; j < 8; ++j) {
      v16h b = bfrag(Bt + (j * 16) * 40, 40);
      acc[j] = wmma16(a, b, acc[j]);
    }
    __syncthreads();
  }

  const int col = lane & 15, hi = lane >> 4;
#pragma unroll
  for (int j = 0; j < 8; ++j)
#pragma unroll
    for (int r = 0; r < 8; ++r) {
      int row = mb0 + wid * 16 + hi * 8 + r;
      int gl  = nb0 + j * 16 + col;
      out[(size_t)row * L_LEN + gl] = (acc[j][r] + bo[row]) * mask[gl];
    }
}

// ---------------------------------------------------------------------------
extern "C" void kernel_launch(void* const* d_in, const int* in_sizes, int n_in,
                              void* d_out, int out_size, void* d_ws, size_t ws_size,
                              hipStream_t stream) {
  const float* x1   = (const float*)d_in[0];
  // d_in[1] = x2 (unused by reference)
  const float* mask = (const float*)d_in[2];
  const float* Wq   = (const float*)d_in[3];
  const float* bq   = (const float*)d_in[4];
  const float* Wk   = (const float*)d_in[5];
  const float* bk   = (const float*)d_in[6];
  const float* Wv   = (const float*)d_in[7];
  const float* bv   = (const float*)d_in[8];
  const float* Wo   = (const float*)d_in[9];
  const float* bo   = (const float*)d_in[10];
  float* out = (float*)d_out;

  char* ws = (char*)d_ws;
  const size_t QKT_BYTES = (size_t)L_LEN * QDIM * sizeof(h16);       // 32 MiB
  const size_t VH_BYTES  = (size_t)CDIM * L_LEN * sizeof(h16);       // 16 MiB
  const size_t EN_BYTES  = (size_t)NB * BLQ * WIN * sizeof(float);   // 128 MiB
  const size_t AT_BYTES  = (size_t)NB * BLQ * WIN * sizeof(h16);     // 64 MiB
  h16*   qkT  = (h16*)ws;
  h16*   vh   = (h16*)(ws + QKT_BYTES);
  float* ener = (float*)(ws + QKT_BYTES + VH_BYTES);
  h16*   attn = (h16*)(ws + QKT_BYTES + VH_BYTES + EN_BYTES);
  h16*   uT   = (h16*)(ws + QKT_BYTES + VH_BYTES + EN_BYTES + AT_BYTES);

  k1_qkv    <<<dim3(L_LEN / 128, 6), 256, 0, stream>>>(x1, Wq, bq, Wk, bk, Wv, bv, qkT, vh);
  k2_energy <<<dim3(8, NB),          256, 0, stream>>>(qkT, ener);
  k3_softmax<<<dim3(NB * BLQ / 8),   256, 0, stream>>>(ener, mask, attn);
  k4_attout <<<dim3(4, 2, NB),       256, 0, stream>>>(vh, attn, uT);
  k5_outproj<<<dim3(L_LEN / 128, 4), 256, 0, stream>>>(uT, Wo, bo, mask, out);
}
